// SkillsEncoder_37976100831406
// MI455X (gfx1250) — matrix operand order repaired and checked by reading, not compiled, not run on hardware
//
#include <hip/hip_runtime.h>
#include <hip/hip_bf16.h>

typedef __attribute__((ext_vector_type(16))) __bf16 v16bf;
typedef __attribute__((ext_vector_type(8)))  float  v8f;

namespace {
constexpr int kB = 256, kS = 20, kL = 512, kD = 128, kH = 768;
constexpr int kSP    = 32;   // S padded to 2 WMMA M-tiles
constexpr int kWaves = 8;    // 256 threads / wave32

// Dynamic LDS layout (strides padded against bank conflicts)
constexpr int SEB_STRIDE = kD + 8;   // bf16 elems
constexpr int Q_STRIDE   = kH + 8;   // bf16 elems
constexpr int SC_STRIDE  = kL + 1;   // f32  elems
constexpr int AT_STRIDE  = kL + 8;   // bf16 elems
constexpr size_t OFF_SEB = 0;
constexpr size_t OFF_Q   = OFF_SEB + (size_t)kSP * SEB_STRIDE * sizeof(__bf16);
constexpr size_t OFF_SC  = OFF_Q   + (size_t)kSP * Q_STRIDE   * sizeof(__bf16);
constexpr size_t OFF_AT  = OFF_SC  + (size_t)kSP * SC_STRIDE  * sizeof(float);
constexpr size_t SMEM_BYTES = OFF_AT + (size_t)kSP * AT_STRIDE * sizeof(__bf16);
} // namespace

__device__ __forceinline__ float waveMax(float v) {
#pragma unroll
  for (int off = 16; off; off >>= 1) v = fmaxf(v, __shfl_xor(v, off, 32));
  return v;
}
__device__ __forceinline__ float waveSum(float v) {
#pragma unroll
  for (int off = 16; off; off >>= 1) v += __shfl_xor(v, off, 32);
  return v;
}
__device__ __forceinline__ v8f wmma_bf16(v16bf a, v16bf b, v8f c) {
  // D = A(16x32 bf16) * B(32x16 bf16) + C(16x16 f32)
  return __builtin_amdgcn_wmma_f32_16x16x32_bf16(false, a, false, b,
                                                 (short)0, c, false, false);
}

__global__ __launch_bounds__(256, 1)
void skills_encoder_kernel(const int*   __restrict__ skills,
                           const float* __restrict__ desc,
                           const float* __restrict__ embs,
                           const float* __restrict__ W,
                           float*       __restrict__ out) {
  extern __shared__ char smem[];
  __bf16* sSE = (__bf16*)(smem + OFF_SEB);  // [kSP][SEB_STRIDE]
  __bf16* sQ  = (__bf16*)(smem + OFF_Q);    // [kSP][Q_STRIDE]
  float*  sSc = (float*) (smem + OFF_SC);   // [kSP][SC_STRIDE]
  __bf16* sAt = (__bf16*)(smem + OFF_AT);   // [kSP][AT_STRIDE]

  const int b    = blockIdx.x;
  const int tid  = threadIdx.x;
  const int wave = tid >> 5;
  const int lane = tid & 31;
  const int half = lane >> 4;   // lane group 0/1
  const int l16  = lane & 15;
  const int kbA  = half * 16;   // 16-bit A-frag K base (contiguous, matches B)
  const int kbB  = half * 16;   // 16-bit B-frag K base
  const int hi8  = half * 8;    // f32 C/D row offset

  // ---- Phase 0: gather skill embeddings -> LDS bf16, rows >= kS zeroed ----
  for (int idx = tid; idx < kSP * kD; idx += blockDim.x) {
    const int s = idx / kD, d = idx - s * kD;
    float v = 0.0f;
    if (s < kS) v = embs[(long)skills[b * kS + s] * kD + d];
    sSE[s * SEB_STRIDE + d] = (__bf16)v;
  }
  __syncthreads();

  // ---- Phase 1: q = se @ W  -> sQ (32 x 768 bf16) ----
  for (int nt = wave; nt < kH / 16; nt += kWaves) {
    const int nb = nt * 16;
    v8f c0 = {}, c1 = {};
#pragma unroll
    for (int ks = 0; ks < kD / 32; ++ks) {
      v16bf a0, a1, bm;
#pragma unroll
      for (int h = 0; h < 16; ++h) {
        const int k = ks * 32 + kbA + h;
        a0[h] = sSE[l16 * SEB_STRIDE + k];
        a1[h] = sSE[(16 + l16) * SEB_STRIDE + k];
        bm[h] = (__bf16)W[(long)(ks * 32 + kbB + h) * kH + nb + l16];
      }
      c0 = wmma_bf16(a0, bm, c0);           // B-fragment reused for both M-tiles
      c1 = wmma_bf16(a1, bm, c1);
    }
#pragma unroll
    for (int r = 0; r < 8; ++r) {
      sQ[(r + hi8) * Q_STRIDE + nb + l16]      = (__bf16)c0[r];
      sQ[(16 + r + hi8) * Q_STRIDE + nb + l16] = (__bf16)c1[r];
    }
  }
  __syncthreads();

  // ---- Phase 2: scores = q @ desc^T -> sSc (32 x 512 f32) ----
  const float* descB = desc + (long)b * kL * kH;
  for (int nt = wave; nt < kL / 16; nt += kWaves) {
    const int lb = nt * 16;
    v8f c0 = {}, c1 = {};
    const float* dcol = descB + (long)(lb + l16) * kH + kbB;  // 16 contiguous f32 per frag
    for (int ks = 0; ks < kH / 32; ++ks) {
      v16bf a0, a1, bm;
#pragma unroll
      for (int h = 0; h < 16; ++h) {
        const int k = ks * 32 + kbA + h;
        a0[h] = sQ[l16 * Q_STRIDE + k];
        a1[h] = sQ[(16 + l16) * Q_STRIDE + k];
        bm[h] = (__bf16)dcol[ks * 32 + h];
      }
      c0 = wmma_bf16(a0, bm, c0);
      c1 = wmma_bf16(a1, bm, c1);
    }
#pragma unroll
    for (int r = 0; r < 8; ++r) {
      sSc[(r + hi8) * SC_STRIDE + lb + l16]      = c0[r];
      sSc[(16 + r + hi8) * SC_STRIDE + lb + l16] = c1[r];
    }
  }
  __syncthreads();

  // ---- Phase 3: row softmax over L (wave-per-row, wave32 shuffles) ----
  for (int r = wave; r < kS; r += kWaves) {
    float vals[16], vmax = -3.0e38f;
#pragma unroll
    for (int i = 0; i < 16; ++i) {
      vals[i] = sSc[r * SC_STRIDE + lane + 32 * i];
      vmax = fmaxf(vmax, vals[i]);
    }
    vmax = waveMax(vmax);
    float sum = 0.0f;
#pragma unroll
    for (int i = 0; i < 16; ++i) { vals[i] = __expf(vals[i] - vmax); sum += vals[i]; }
    const float inv = 1.0f / waveSum(sum);
#pragma unroll
    for (int i = 0; i < 16; ++i)
      sAt[r * AT_STRIDE + lane + 32 * i] = (__bf16)(vals[i] * inv);
  }
  // Zero padded attention rows so pass-4 A tiles are clean
  for (int idx = tid; idx < (kSP - kS) * kL; idx += blockDim.x) {
    const int rr = kS + idx / kL, cc = idx - (idx / kL) * kL;
    sAt[rr * AT_STRIDE + cc] = (__bf16)0.0f;
  }
  __syncthreads();

  // ---- Phase 4: out = attn @ desc -> global (concat slot D..D+H) ----
  for (int nt = wave; nt < kH / 16; nt += kWaves) {
    const int hb = nt * 16;
    v8f c0 = {}, c1 = {};
    for (int ks = 0; ks < kL / 32; ++ks) {
      if (ks + 1 < kL / 32)
        __builtin_prefetch(descB + (long)((ks + 1) * 32 + kbB) * kH + hb + l16, 0, 0);
      v16bf a0, a1, bm;
#pragma unroll
      for (int h = 0; h < 16; ++h) {
        const int k = ks * 32 + kbA + h;
        a0[h] = sAt[l16 * AT_STRIDE + k];
        a1[h] = sAt[(16 + l16) * AT_STRIDE + k];
        bm[h] = (__bf16)descB[(long)(ks * 32 + kbB + h) * kH + hb + l16];
      }
      c0 = wmma_bf16(a0, bm, c0);
      c1 = wmma_bf16(a1, bm, c1);
    }
#pragma unroll
    for (int r = 0; r < 8; ++r) {
      const int s0 = r + hi8;                       // rows 0..15, all valid
      out[((long)b * kS + s0) * (kD + kH) + kD + hb + l16] = c0[r];
      const int s1 = 16 + r + hi8;                  // rows 16..31
      if (s1 < kS)
        out[((long)b * kS + s1) * (kD + kH) + kD + hb + l16] = c1[r];
    }
  }

  // ---- Epilogue: copy se (f32) into concat slot 0..D ----
  for (int idx = tid; idx < kS * kD; idx += blockDim.x) {
    const int s = idx / kD, d = idx - s * kD;
    out[((long)b * kS + s) * (kD + kH) + d] =
        embs[(long)skills[b * kS + s] * kD + d];
  }
}

extern "C" void kernel_launch(void* const* d_in, const int* in_sizes, int n_in,
                              void* d_out, int out_size, void* d_ws, size_t ws_size,
                              hipStream_t stream) {
  const int*   skills = (const int*)  d_in[0];
  const float* desc   = (const float*)d_in[1];
  const float* embs   = (const float*)d_in[2];
  const float* W      = (const float*)d_in[3];
  float* out = (float*)d_out;

  hipFuncSetAttribute((const void*)skills_encoder_kernel,
                      hipFuncAttributeMaxDynamicSharedMemorySize,
                      (int)SMEM_BYTES);
  skills_encoder_kernel<<<dim3(kB), dim3(256), SMEM_BYTES, stream>>>(
      skills, desc, embs, W, out);
}
